// CrossAttnetion_all_88184268521591
// MI455X (gfx1250) — matrix operand order repaired and checked by reading, not compiled, not run on hardware
//
#include <hip/hip_runtime.h>
#include <hip/hip_bf16.h>

#define NPTS 8192
#define CCH  256
#define HEADS 4
#define DHEAD 64
#define EPSV 1e-5f

typedef __bf16 bf16_t;
typedef __attribute__((ext_vector_type(16))) __bf16 v16bf;
typedef __attribute__((ext_vector_type(8)))  __bf16 v8bf;
typedef __attribute__((ext_vector_type(8)))  float  v8f;

// ---------------------------------------------------------------------------
// CDNA5 async global->LDS copies (ASYNCcnt-tracked, bypass VGPRs)
// signature: (v4i AS1* gsrc, v4i AS3* ldsdst, imm offset, imm cpol)
// ---------------------------------------------------------------------------
#if __has_builtin(__builtin_amdgcn_global_load_async_to_lds_b128)
#define HAVE_ASYNC 1
typedef int v4i __attribute__((vector_size(4 * sizeof(int))));
typedef __attribute__((address_space(1))) v4i gv4i;
typedef __attribute__((address_space(3))) v4i lv4i;
static __device__ inline void async_cp16(const bf16_t* g, bf16_t* l) {
  __builtin_amdgcn_global_load_async_to_lds_b128((gv4i*)g, (lv4i*)l, 0, 0);
}
static __device__ inline void wait_async0() {
  asm volatile("s_wait_asynccnt 0x0" ::: "memory");
}
#else
#define HAVE_ASYNC 0
#endif

// ---------------------------------------------------------------------------
// WMMA helpers (CDNA5 wave32, V_WMMA_F32_16X16X32_BF16)
// ---------------------------------------------------------------------------
static __device__ inline v8f wmma_bf16(v16bf a, v16bf b, v8f c) {
  return __builtin_amdgcn_wmma_f32_16x16x32_bf16(false, a, false, b, (short)0, c,
                                                 false, false);
}

static __device__ inline v16bf cat8(v8bf lo, v8bf hi) {
  return __builtin_shufflevector(lo, hi, 0, 1, 2, 3, 4, 5, 6, 7, 8, 9, 10, 11,
                                 12, 13, 14, 15);
}

// A-matrix 16x32 bf16 fragment from row-major [16 x stride] source.
// lane 0-15: M=lane, K chunks {0..7,16..23}; lanes 16-31: {8..15,24..31}
// -> two contiguous 16B loads per lane (ds_load_b128 / global_load_b128)
static __device__ inline v16bf load_a_frag(const bf16_t* base, int stride) {
  const int lane = threadIdx.x & 31;
  const bf16_t* p = base + (lane & 15) * stride + (lane >> 4) * 8;
  const v8bf lo = *(const v8bf*)(p);
  const v8bf hi = *(const v8bf*)(p + 16);
  return cat8(lo, hi);
}

// B-matrix 32x16 fragment from a K-contiguous ("transposed") [16 x stride]
// source: row n holds the 32 K-values of output column n.
static __device__ inline v16bf load_b_fragT(const bf16_t* base, int stride) {
  const int lane = threadIdx.x & 31;
  const bf16_t* p = base + (lane & 15) * stride + (lane >> 4) * 8;
  const v8bf lo = *(const v8bf*)(p);
  const v8bf hi = *(const v8bf*)(p + 16);
  return cat8(lo, hi);
}

static __device__ inline void zero_acc(v8f* acc, int n) {
  for (int j = 0; j < n; ++j)
#pragma unroll
    for (int e = 0; e < 8; ++e) acc[j][e] = 0.0f;
}

// reductions across the 16 lanes of a half-wave (masks < 16 stay in-half)
static __device__ inline float redmax16(float v) {
  v = fmaxf(v, __shfl_xor(v, 1, 32));
  v = fmaxf(v, __shfl_xor(v, 2, 32));
  v = fmaxf(v, __shfl_xor(v, 4, 32));
  v = fmaxf(v, __shfl_xor(v, 8, 32));
  return v;
}
static __device__ inline float redsum16(float v) {
  v += __shfl_xor(v, 1, 32);
  v += __shfl_xor(v, 2, 32);
  v += __shfl_xor(v, 4, 32);
  v += __shfl_xor(v, 8, 32);
  return v;
}

// ---------------------------------------------------------------------------
// descs fp32 -> bf16 copy (A operand for the concat GEMM)
// ---------------------------------------------------------------------------
__global__ void k_convert_x(const float* __restrict__ descs, bf16_t* __restrict__ Xb) {
  const int i = blockIdx.x * blockDim.x + threadIdx.x;
  Xb[i] = (bf16_t)descs[i];
}

// ---------------------------------------------------------------------------
// QKV projection: 128x64 output tile per block (128 thr = 4 waves, each wave
// 32 rows x 64 cols = 8 WMMA / K-step sharing 4 B fragments).
// Q stored [h][n][d] pre-scaled by 1/sqrt(DIM); K stored [h][m][d];
// V stored transposed [h][d][m]. channel c = d*H + h
// ---------------------------------------------------------------------------
__global__ void k_qkv(const float* __restrict__ descs,
                      const float* __restrict__ Wq, const float* __restrict__ bq,
                      const float* __restrict__ Wk, const float* __restrict__ bk,
                      const float* __restrict__ Wv, const float* __restrict__ bv,
                      bf16_t* __restrict__ Qh, bf16_t* __restrict__ Kh,
                      bf16_t* __restrict__ Vt) {
  __shared__ __align__(32) bf16_t At[128 * 32];  // [row][k]
  __shared__ __align__(32) bf16_t Bt[64 * 32];   // [o][k]  (K-contiguous)
  const int n0 = blockIdx.x * 128;
  const int o0 = blockIdx.y * 64;
  const int mat = blockIdx.z;
  const float* W = (mat == 0) ? Wq : (mat == 1) ? Wk : Wv;
  const float* bias = (mat == 0) ? bq : (mat == 1) ? bk : bv;
  const int tid = threadIdx.x, wave = tid >> 5, lane = tid & 31;

  v8f acc[8];
  zero_acc(acc, 8);

  for (int k0 = 0; k0 < CCH; k0 += 32) {
    for (int idx = tid; idx < 128 * 32; idx += 128) {
      const int r = idx >> 5, k = idx & 31;
      At[idx] = (bf16_t)descs[(size_t)(n0 + r) * CCH + (k0 + k)];
    }
    for (int idx = tid; idx < 64 * 32; idx += 128) {
      const int o = idx >> 5, k = idx & 31;
      Bt[idx] = (bf16_t)W[(size_t)(o0 + o) * CCH + (k0 + k)];
    }
    if (k0 + 32 < CCH) __builtin_prefetch(&W[(size_t)o0 * CCH + k0 + 32], 0, 3);
    __syncthreads();
    const v16bf a0 = load_a_frag(At + (wave * 32) * 32, 32);
    const v16bf a1 = load_a_frag(At + (wave * 32 + 16) * 32, 32);
#pragma unroll
    for (int j = 0; j < 4; ++j) {
      const v16bf b = load_b_fragT(Bt + (j * 16) * 32, 32);
      acc[j] = wmma_bf16(a0, b, acc[j]);
      acc[4 + j] = wmma_bf16(a1, b, acc[4 + j]);
    }
    __syncthreads();
  }

  const int colbase = lane & 15, rbase = (lane >> 4) * 8;
#pragma unroll
  for (int g = 0; g < 2; ++g) {
#pragma unroll
    for (int j = 0; j < 4; ++j) {
      const int o = o0 + j * 16 + colbase;
      const float bo = bias[o];
      const int h = o & (HEADS - 1);
      const int d = o >> 2;
#pragma unroll
      for (int r = 0; r < 8; ++r) {
        const int np = n0 + wave * 32 + g * 16 + rbase + r;
        const float v = acc[g * 4 + j][r] + bo;
        if (mat == 0)
          Qh[((size_t)h * NPTS + np) * DHEAD + d] = (bf16_t)(v * 0.125f);
        else if (mat == 1)
          Kh[((size_t)h * NPTS + np) * DHEAD + d] = (bf16_t)v;
        else
          Vt[((size_t)h * DHEAD + d) * NPTS + np] = (bf16_t)v;
      }
    }
  }
}

// ---------------------------------------------------------------------------
// Flash attention per head. 256 thr = 8 waves, each wave owns 16 Q rows.
// Ktile [m][d] (d-contiguous for QK^T B-frags), Vtile [d][m] (m-contiguous
// for PV B-frags). With async-LDS support: double-buffered tiles streamed by
// GLOBAL_LOAD_ASYNC_TO_LDS_B128, prefetching tile i+1 while computing tile i.
// ---------------------------------------------------------------------------
__global__ void k_attn(const bf16_t* __restrict__ Qh, const bf16_t* __restrict__ Kh,
                       const bf16_t* __restrict__ Vt, bf16_t* __restrict__ msgb) {
#if HAVE_ASYNC
  __shared__ __align__(32) bf16_t Ktile[2][64 * 64];  // [m][d]
  __shared__ __align__(32) bf16_t Vtile[2][64 * 64];  // [d][m]
#else
  __shared__ __align__(32) bf16_t Ktile[1][64 * 64];
  __shared__ __align__(32) bf16_t Vtile[1][64 * 64];
#endif
  __shared__ __align__(32) bf16_t Pbuf[8 * 16 * 64];  // per-wave [row][m]
  const int h = blockIdx.y;
  const int n0 = blockIdx.x * 128;
  const int tid = threadIdx.x, wave = tid >> 5, lane = tid & 31;
  const int colbase = lane & 15, rbase = (lane >> 4) * 8;

  const bf16_t* Qbase = Qh + ((size_t)h * NPTS + n0 + wave * 16) * DHEAD;
  const v16bf q0 = load_a_frag(Qbase, DHEAD);
  const v16bf q1 = load_a_frag(Qbase + 32, DHEAD);

  v8f O[4];
  zero_acc(O, 4);
  float mrow[8], lrow[8];
#pragma unroll
  for (int r = 0; r < 8; ++r) { mrow[r] = -1e30f; lrow[r] = 0.0f; }

  bf16_t* Pw = Pbuf + wave * 16 * 64;
  const int ITERS = NPTS / 64;

#if HAVE_ASYNC
  // prologue: stage tile 0 into buffer 0 (512 x 16B chunks per tile)
  for (int idx = tid; idx < 512; idx += 256) {
    const int m = idx >> 3, c = (idx & 7) * 8;
    async_cp16(&Kh[((size_t)h * NPTS + m) * DHEAD + c], &Ktile[0][m * 64 + c]);
  }
  for (int idx = tid; idx < 512; idx += 256) {
    const int d = idx >> 3, c = (idx & 7) * 8;
    async_cp16(&Vt[((size_t)h * DHEAD + d) * NPTS + c], &Vtile[0][d * 64 + c]);
  }
#endif

  for (int it = 0; it < ITERS; ++it) {
    const int m0 = it * 64;
#if HAVE_ASYNC
    const int cur = it & 1;
    wait_async0();
    __syncthreads();  // tile(cur) complete for all waves; buf(cur^1) free
    if (it + 1 < ITERS) {
      const int m1 = m0 + 64;
      for (int idx = tid; idx < 512; idx += 256) {
        const int m = idx >> 3, c = (idx & 7) * 8;
        async_cp16(&Kh[((size_t)h * NPTS + m1 + m) * DHEAD + c],
                   &Ktile[cur ^ 1][m * 64 + c]);
      }
      for (int idx = tid; idx < 512; idx += 256) {
        const int d = idx >> 3, c = (idx & 7) * 8;
        async_cp16(&Vt[((size_t)h * DHEAD + d) * NPTS + m1 + c],
                   &Vtile[cur ^ 1][d * 64 + c]);
      }
    }
    const bf16_t* Kt_ = Ktile[cur];
    const bf16_t* Vt_ = Vtile[cur];
#else
    for (int idx = tid; idx < 64 * 64; idx += 256) {
      const int m = idx >> 6, d = idx & 63;
      Ktile[0][idx] = Kh[((size_t)h * NPTS + m0 + m) * DHEAD + d];
    }
    for (int idx = tid; idx < 64 * 64; idx += 256) {
      const int d = idx >> 6, m = idx & 63;
      Vtile[0][idx] = Vt[((size_t)h * DHEAD + d) * NPTS + m0 + m];
    }
    __syncthreads();
    const bf16_t* Kt_ = Ktile[0];
    const bf16_t* Vt_ = Vtile[0];
#endif

    // S = Q K^T (scale folded into Q), 16x64 per wave
    v8f S[4];
    zero_acc(S, 4);
#pragma unroll
    for (int j = 0; j < 4; ++j) {
      S[j] = wmma_bf16(q0, load_b_fragT(Kt_ + (j * 16) * 64, 64), S[j]);
      S[j] = wmma_bf16(q1, load_b_fragT(Kt_ + (j * 16) * 64 + 32, 64), S[j]);
    }

    // online softmax update (rows rbase..rbase+7 of this wave tile)
#pragma unroll
    for (int r = 0; r < 8; ++r) {
      float v = fmaxf(fmaxf(S[0][r], S[1][r]), fmaxf(S[2][r], S[3][r]));
      const float rmax = redmax16(v);
      const float mnew = fmaxf(mrow[r], rmax);
      const float cf = __expf(mrow[r] - mnew);
      float psum = 0.0f;
#pragma unroll
      for (int j = 0; j < 4; ++j) {
        const float p = __expf(S[j][r] - mnew);
        S[j][r] = p;
        psum += p;
      }
      psum = redsum16(psum);
      lrow[r] = lrow[r] * cf + psum;
      mrow[r] = mnew;
#pragma unroll
      for (int jd = 0; jd < 4; ++jd) O[jd][r] *= cf;
    }

    // stage P into per-wave LDS region, row-major [16][64]
#pragma unroll
    for (int j = 0; j < 4; ++j)
#pragma unroll
      for (int r = 0; r < 8; ++r)
        Pw[(rbase + r) * 64 + j * 16 + colbase] = (bf16_t)S[j][r];

    // O += P @ V
    const v16bf p0 = load_a_frag(Pw, 64);
    const v16bf p1 = load_a_frag(Pw + 32, 64);
#pragma unroll
    for (int jd = 0; jd < 4; ++jd) {
      O[jd] = wmma_bf16(p0, load_b_fragT(Vt_ + (jd * 16) * 64, 64), O[jd]);
      O[jd] = wmma_bf16(p1, load_b_fragT(Vt_ + (jd * 16) * 64 + 32, 64), O[jd]);
    }
#if !HAVE_ASYNC
    __syncthreads();
#endif
  }

  // epilogue: normalize and scatter to msgb[n][c], c = d*H + h
#pragma unroll
  for (int jd = 0; jd < 4; ++jd) {
    const int d = jd * 16 + colbase;
    const int c = d * HEADS + h;
#pragma unroll
    for (int r = 0; r < 8; ++r) {
      const int np = n0 + wave * 16 + rbase + r;
      msgb[(size_t)np * CCH + c] = (bf16_t)(O[jd][r] / lrow[r]);
    }
  }
}

// ---------------------------------------------------------------------------
// m2 = msgb @ Wm^T + bm  (bf16 out, K = 256); 128x64 tile per block.
// A tile (already bf16) streamed with async global->LDS copies when available.
// ---------------------------------------------------------------------------
__global__ void k_gemm_m(const bf16_t* __restrict__ A, const float* __restrict__ W,
                         const float* __restrict__ bias, bf16_t* __restrict__ Out) {
  __shared__ __align__(32) bf16_t At[128 * 32];
  __shared__ __align__(32) bf16_t Bt[64 * 32];
  const int n0 = blockIdx.x * 128;
  const int o0 = blockIdx.y * 64;
  const int tid = threadIdx.x, wave = tid >> 5, lane = tid & 31;
  v8f acc[8];
  zero_acc(acc, 8);

  for (int k0 = 0; k0 < CCH; k0 += 32) {
#if HAVE_ASYNC
    for (int idx = tid; idx < 512; idx += 128) {
      const int r = idx >> 2, c = (idx & 3) * 8;
      async_cp16(&A[(size_t)(n0 + r) * CCH + k0 + c], &At[r * 32 + c]);
    }
#else
    for (int idx = tid; idx < 128 * 32; idx += 128) {
      const int r = idx >> 5, k = idx & 31;
      At[idx] = A[(size_t)(n0 + r) * CCH + (k0 + k)];
    }
#endif
    for (int idx = tid; idx < 64 * 32; idx += 128) {
      const int o = idx >> 5, k = idx & 31;
      Bt[idx] = (bf16_t)W[(size_t)(o0 + o) * CCH + (k0 + k)];
    }
    if (k0 + 32 < CCH) __builtin_prefetch(&W[(size_t)o0 * CCH + k0 + 32], 0, 3);
#if HAVE_ASYNC
    wait_async0();
#endif
    __syncthreads();
    const v16bf a0 = load_a_frag(At + (wave * 32) * 32, 32);
    const v16bf a1 = load_a_frag(At + (wave * 32 + 16) * 32, 32);
#pragma unroll
    for (int j = 0; j < 4; ++j) {
      const v16bf b = load_b_fragT(Bt + (j * 16) * 32, 32);
      acc[j] = wmma_bf16(a0, b, acc[j]);
      acc[4 + j] = wmma_bf16(a1, b, acc[4 + j]);
    }
    __syncthreads();
  }

  const int colbase = lane & 15, rbase = (lane >> 4) * 8;
#pragma unroll
  for (int g = 0; g < 2; ++g)
#pragma unroll
    for (int j = 0; j < 4; ++j) {
      const int o = o0 + j * 16 + colbase;
      const float bo = bias[o];
#pragma unroll
      for (int r = 0; r < 8; ++r) {
        const int np = n0 + wave * 32 + g * 16 + rbase + r;
        Out[(size_t)np * CCH + o] = (bf16_t)(acc[g * 4 + j][r] + bo);
      }
    }
}

// ---------------------------------------------------------------------------
// y1 = W1 @ concat(x, m2) + b1 (fp32 out, K = 512, O = 512)
// accumulates per-channel sum / sumsq for InstanceNorm
// ---------------------------------------------------------------------------
__global__ void k_gemm1(const bf16_t* __restrict__ Xb, const bf16_t* __restrict__ M2,
                        const float* __restrict__ W1, const float* __restrict__ b1,
                        float* __restrict__ Y1, float* __restrict__ ssum,
                        float* __restrict__ ssq) {
  __shared__ __align__(32) bf16_t At[128 * 32];
  __shared__ __align__(32) bf16_t Bt[64 * 32];
  const int n0 = blockIdx.x * 128;
  const int o0 = blockIdx.y * 64;
  const int tid = threadIdx.x, wave = tid >> 5, lane = tid & 31;
  v8f acc[8];
  zero_acc(acc, 8);

  for (int k0 = 0; k0 < 2 * CCH; k0 += 32) {
    const bf16_t* Asrc = (k0 < CCH) ? Xb : M2;
    const int kk0 = (k0 < CCH) ? k0 : (k0 - CCH);
#if HAVE_ASYNC
    for (int idx = tid; idx < 512; idx += 128) {
      const int r = idx >> 2, c = (idx & 3) * 8;
      async_cp16(&Asrc[(size_t)(n0 + r) * CCH + kk0 + c], &At[r * 32 + c]);
    }
#else
    for (int idx = tid; idx < 128 * 32; idx += 128) {
      const int r = idx >> 5, k = idx & 31;
      At[idx] = Asrc[(size_t)(n0 + r) * CCH + (kk0 + k)];
    }
#endif
    for (int idx = tid; idx < 64 * 32; idx += 128) {
      const int o = idx >> 5, k = idx & 31;
      Bt[idx] = (bf16_t)W1[(size_t)(o0 + o) * (2 * CCH) + (k0 + k)];
    }
    if (k0 + 32 < 2 * CCH)
      __builtin_prefetch(&W1[(size_t)o0 * (2 * CCH) + k0 + 32], 0, 3);
#if HAVE_ASYNC
    wait_async0();
#endif
    __syncthreads();
    const v16bf a0 = load_a_frag(At + (wave * 32) * 32, 32);
    const v16bf a1 = load_a_frag(At + (wave * 32 + 16) * 32, 32);
#pragma unroll
    for (int j = 0; j < 4; ++j) {
      const v16bf b = load_b_fragT(Bt + (j * 16) * 32, 32);
      acc[j] = wmma_bf16(a0, b, acc[j]);
      acc[4 + j] = wmma_bf16(a1, b, acc[4 + j]);
    }
    __syncthreads();
  }

  const int colbase = lane & 15, rbase = (lane >> 4) * 8;
#pragma unroll
  for (int g = 0; g < 2; ++g)
#pragma unroll
    for (int j = 0; j < 4; ++j) {
      const int o = o0 + j * 16 + colbase;
      const float bo = b1[o];
      float s = 0.0f, q = 0.0f;
#pragma unroll
      for (int r = 0; r < 8; ++r) {
        const int np = n0 + wave * 32 + g * 16 + rbase + r;
        const float v = acc[g * 4 + j][r] + bo;
        Y1[(size_t)np * (2 * CCH) + o] = v;
        s += v;
        q += v * v;
      }
      atomicAdd(&ssum[o], s);
      atomicAdd(&ssq[o], q);
    }
}

// ---------------------------------------------------------------------------
// InstanceNorm statistics
// ---------------------------------------------------------------------------
__global__ void k_zero(float* __restrict__ p) { p[threadIdx.x] = 0.0f; }

__global__ void k_stats(const float* __restrict__ ssum, const float* __restrict__ ssq,
                        float* __restrict__ mu, float* __restrict__ rinv) {
  const int c = threadIdx.x;
  const float inv_n = 1.0f / (float)NPTS;
  const float m = ssum[c] * inv_n;
  const float var = ssq[c] * inv_n - m * m;
  mu[c] = m;
  rinv[c] = rsqrtf(var + EPSV);
}

// ---------------------------------------------------------------------------
// out[n][c] = descs[n][c] + W2 @ relu(norm(y1)) + b2  (K = 512, O = 256)
// norm+relu applied on the fly while staging A into LDS
// ---------------------------------------------------------------------------
__global__ void k_gemm2(const float* __restrict__ Y1, const float* __restrict__ mu,
                        const float* __restrict__ rinv, const float* __restrict__ W2,
                        const float* __restrict__ b2, const float* __restrict__ descs,
                        float* __restrict__ Out) {
  __shared__ __align__(32) bf16_t At[128 * 32];
  __shared__ __align__(32) bf16_t Bt[64 * 32];
  const int n0 = blockIdx.x * 128;
  const int o0 = blockIdx.y * 64;
  const int tid = threadIdx.x, wave = tid >> 5, lane = tid & 31;
  v8f acc[8];
  zero_acc(acc, 8);

  for (int k0 = 0; k0 < 2 * CCH; k0 += 32) {
    for (int idx = tid; idx < 128 * 32; idx += 128) {
      const int r = idx >> 5, k = idx & 31;
      const int kk = k0 + k;
      float t = (Y1[(size_t)(n0 + r) * (2 * CCH) + kk] - mu[kk]) * rinv[kk];
      At[idx] = (bf16_t)fmaxf(t, 0.0f);
    }
    for (int idx = tid; idx < 64 * 32; idx += 128) {
      const int o = idx >> 5, k = idx & 31;
      Bt[idx] = (bf16_t)W2[(size_t)(o0 + o) * (2 * CCH) + (k0 + k)];
    }
    if (k0 + 32 < 2 * CCH)
      __builtin_prefetch(&W2[(size_t)o0 * (2 * CCH) + k0 + 32], 0, 3);
    __syncthreads();
    const v16bf a0 = load_a_frag(At + (wave * 32) * 32, 32);
    const v16bf a1 = load_a_frag(At + (wave * 32 + 16) * 32, 32);
#pragma unroll
    for (int j = 0; j < 4; ++j) {
      const v16bf b = load_b_fragT(Bt + (j * 16) * 32, 32);
      acc[j] = wmma_bf16(a0, b, acc[j]);
      acc[4 + j] = wmma_bf16(a1, b, acc[4 + j]);
    }
    __syncthreads();
  }

  const int colbase = lane & 15, rbase = (lane >> 4) * 8;
#pragma unroll
  for (int g = 0; g < 2; ++g)
#pragma unroll
    for (int j = 0; j < 4; ++j) {
      const int o = o0 + j * 16 + colbase;
      const float bo = b2[o];
#pragma unroll
      for (int r = 0; r < 8; ++r) {
        const int np = n0 + wave * 32 + g * 16 + rbase + r;
        Out[(size_t)np * CCH + o] =
            descs[(size_t)np * CCH + o] + acc[g * 4 + j][r] + bo;
      }
    }
}

// ---------------------------------------------------------------------------
// host launcher
// ---------------------------------------------------------------------------
extern "C" void kernel_launch(void* const* d_in, const int* in_sizes, int n_in,
                              void* d_out, int out_size, void* d_ws, size_t ws_size,
                              hipStream_t stream) {
  const float* descs = (const float*)d_in[0];
  const float* Wq = (const float*)d_in[1];
  const float* bq = (const float*)d_in[2];
  const float* Wk = (const float*)d_in[3];
  const float* bk = (const float*)d_in[4];
  const float* Wv = (const float*)d_in[5];
  const float* bv = (const float*)d_in[6];
  const float* Wm = (const float*)d_in[7];
  const float* bm = (const float*)d_in[8];
  const float* W1 = (const float*)d_in[9];
  const float* b1 = (const float*)d_in[10];
  const float* W2 = (const float*)d_in[11];
  const float* b2 = (const float*)d_in[12];
  float* out = (float*)d_out;

  char* ws = (char*)d_ws;
  const size_t MB = (size_t)1 << 20;
  bf16_t* Xb   = (bf16_t*)(ws + 0 * MB);    // N*C bf16   (4 MB)
  bf16_t* Qh   = (bf16_t*)(ws + 4 * MB);    // H*N*D bf16 (4 MB)
  bf16_t* Kh   = (bf16_t*)(ws + 8 * MB);    // H*N*D bf16 (4 MB)
  bf16_t* Vt   = (bf16_t*)(ws + 12 * MB);   // H*D*N bf16 (4 MB)
  bf16_t* msgb = (bf16_t*)(ws + 16 * MB);   // N*C bf16   (4 MB)
  bf16_t* m2   = (bf16_t*)(ws + 20 * MB);   // N*C bf16   (4 MB)
  float*  y1   = (float*)(ws + 24 * MB);    // N*512 f32  (16 MB)
  float*  ssum = (float*)(ws + 40 * MB);    // 512
  float*  ssq  = ssum + 512;
  float*  mu   = ssum + 1024;
  float*  rinv = ssum + 1536;

  k_zero<<<1, 1024, 0, stream>>>(ssum);  // zeroes ssum+ssq (contiguous)
  k_convert_x<<<(NPTS * CCH) / 256, 256, 0, stream>>>(descs, Xb);
  k_qkv<<<dim3(NPTS / 128, CCH / 64, 3), 128, 0, stream>>>(
      descs, Wq, bq, Wk, bk, Wv, bv, Qh, Kh, Vt);
  k_attn<<<dim3(NPTS / 128, HEADS), 256, 0, stream>>>(Qh, Kh, Vt, msgb);
  k_gemm_m<<<dim3(NPTS / 128, CCH / 64), 128, 0, stream>>>(msgb, Wm, bm, m2);
  k_gemm1<<<dim3(NPTS / 128, (2 * CCH) / 64), 128, 0, stream>>>(
      Xb, m2, W1, b1, y1, ssum, ssq);
  k_stats<<<1, 512, 0, stream>>>(ssum, ssq, mu, rinv);
  k_gemm2<<<dim3(NPTS / 128, CCH / 64), 128, 0, stream>>>(
      y1, mu, rinv, W2, b2, descs, out);
}
